// EVAUpsampling_84164179132496
// MI455X (gfx1250) — matrix operand order repaired and checked by reading, not compiled, not run on hardware
//
#include <hip/hip_runtime.h>
#include <hip/hip_bf16.h>

// ---------------- problem constants ----------------
#define BB 2
#define NN 8192
#define CC 32
#define KK 16
#define RR 4
#define HH 512
#define CF 106
#define CFP 128                      // CF padded to 128 (4 k-steps of 32)
#define PT (BB*NN)                   // 16384 points
#define PIX (PT*KK)                  // 262144 pixels
#define PTS_PER_BLOCK 16

// ---------------- workspace layout (bytes) ----------------
#define O_IDX    ((size_t)0)                         // int   [PT][16]            1 MB
#define O_GXYZ   ((size_t)1048576)                   // float [PT][16][3]         3 MB
#define O_FMAT   ((size_t)4194304)                   // f16   [PIX][128]         64 MB
#define O_R16    ((size_t)71303168)                  // f16   [PT*4][128]        16 MB
#define O_W16    ((size_t)88080384)                  // f16   [3][512][128]     384 KB
#define O_SCORES ((size_t)88473600)                  // float [PT][4][16]         4 MB
#define O_LP     ((size_t)92667904)                  // uint  [BB][512][NN]      32 MB
#define ZERO_U4  ((4194304u + 33554432u) / 16u)      // scores+lprime as uint4

typedef _Float16 half16 __attribute__((ext_vector_type(16)));
typedef float    float8 __attribute__((ext_vector_type(8)));

union FragU { half16 h; uint4 q[2]; };

__device__ __forceinline__ half16 load_fragB(const _Float16* p) {
  FragU f;
  f.q[0] = *reinterpret_cast<const uint4*>(p);
  f.q[1] = *reinterpret_cast<const uint4*>(p + 8);    // 16 contiguous halves
  return f.h;
}
__device__ __forceinline__ half16 load_fragA(const _Float16* p) {
  FragU f;
  f.q[0] = *reinterpret_cast<const uint4*>(p);        // K base..base+7
  f.q[1] = *reinterpret_cast<const uint4*>(p + 8);    // K base+8..base+15
  return f.h;
}

// ---------------- zero init (scores + lprime) ----------------
__global__ __launch_bounds__(256) void zero_kernel(uint4* __restrict__ p, unsigned n) {
  uint4 z = make_uint4(0u, 0u, 0u, 0u);
  for (unsigned i = blockIdx.x * 256u + threadIdx.x; i < n; i += gridDim.x * 256u) p[i] = z;
}

// ---------------- weight f32 -> f16 padded [3][512][128] ----------------
__global__ __launch_bounds__(256) void wconvert_kernel(const float* __restrict__ Wg,
                                                       const float* __restrict__ Wh,
                                                       const float* __restrict__ Wl,
                                                       _Float16* __restrict__ w16) {
  int gid = blockIdx.x * 256 + threadIdx.x;            // 3*512*128
  if (gid >= 3 * 512 * 128) return;
  int mat = gid >> 16;
  int hc  = gid & 65535;
  int h   = hc >> 7;
  int c   = hc & 127;
  const float* W = (mat == 0) ? Wg : ((mat == 1) ? Wh : Wl);
  float v = (c < CF) ? W[h * CF + c] : 0.0f;
  w16[gid] = (_Float16)v;
}

// ---------------- KNN: one wave per point, register top-16 + LDS merge ----------------
__global__ __launch_bounds__(256) void knn_kernel(const float* __restrict__ xyz,
                                                  int* __restrict__ idx) {
  __shared__ float sx[256][3];
  __shared__ float md[256][16];
  __shared__ int   mi[256][16];
  const int lane  = threadIdx.x & 31;
  const int wave  = threadIdx.x >> 5;
  const int point = blockIdx.x * 8 + wave;
  const int b     = point >> 13;
  const int n     = point & (NN - 1);
  const float cx = xyz[(size_t)(b * NN + n) * 3 + 0];
  const float cy = xyz[(size_t)(b * NN + n) * 3 + 1];
  const float cz = xyz[(size_t)(b * NN + n) * 3 + 2];
  float bd[16]; int bi[16];
#pragma unroll
  for (int j = 0; j < 16; ++j) { bd[j] = 3.402823e38f; bi[j] = 0; }

  for (int t = 0; t < NN; t += 256) {
    __syncthreads();
    {
      int i = threadIdx.x;
      sx[i][0] = xyz[(size_t)(b * NN + t + i) * 3 + 0];
      sx[i][1] = xyz[(size_t)(b * NN + t + i) * 3 + 1];
      sx[i][2] = xyz[(size_t)(b * NN + t + i) * 3 + 2];
    }
    __syncthreads();
#pragma unroll
    for (int j = 0; j < 8; ++j) {
      int ml = j * 32 + lane;
      int m  = t + ml;
      float dx = sx[ml][0] - cx, dy = sx[ml][1] - cy, dz = sx[ml][2] - cz;
      float d  = dx * dx + dy * dy + dz * dz;
      if (m != n && d < bd[15]) {
        int ii = m;
#pragma unroll
        for (int q = 0; q < 16; ++q) {
          if (d < bd[q]) { float td = bd[q]; int ti = bi[q]; bd[q] = d; bi[q] = ii; d = td; ii = ti; }
        }
      }
    }
  }
  __syncthreads();
#pragma unroll
  for (int j = 0; j < 16; ++j) { md[threadIdx.x][j] = bd[j]; mi[threadIdx.x][j] = bi[j]; }
  __syncthreads();

  const int wbase = wave * 32;
  int ptr = 0, myOut = 0;
  for (int round = 0; round < 16; ++round) {
    float v = (ptr < 16) ? md[wbase + lane][ptr] : 3.402823e38f;
    int wl = lane, wp = ptr;
#pragma unroll
    for (int off = 16; off >= 1; off >>= 1) {
      float ov = __shfl_xor(v, off);
      int   ol = __shfl_xor(wl, off);
      int   op = __shfl_xor(wp, off);
      if (ov < v) { v = ov; wl = ol; wp = op; }
    }
    int wid = mi[wbase + wl][wp];
    if (lane == wl) ptr++;
    if (lane == round) myOut = wid;
  }
  if (lane < 16) idx[(size_t)point * 16 + lane] = myOut;
}

// ---------------- build Fmat16 (pixel-major, CF padded to 128) + grouped_xyz f32 ----------------
__global__ __launch_bounds__(256) void build_kernel(const float* __restrict__ xyz,
                                                    const float* __restrict__ f,
                                                    const int* __restrict__ idx,
                                                    _Float16* __restrict__ fmat16,
                                                    float* __restrict__ gxyz) {
  int gid = blockIdx.x * 256 + threadIdx.x;            // pixel = ((b*N+n)*K + k)
  if (gid >= PIX) return;
  int b  = gid >> 17;                                  // / (N*K)
  int nk = gid & (NN * KK - 1);
  int n  = nk >> 4;
  int m  = idx[gid];
  union { _Float16 h[128]; uint4 q[16]; } fm;
  const float* fb = f + (size_t)b * CC * NN;
#pragma unroll
  for (int c = 0; c < CC; ++c) {
    float gv = fb[c * NN + m];
    float cv = fb[c * NN + n];
    fm.h[c]      = (_Float16)gv;          // grouped_f
    fm.h[32 + c] = (_Float16)cv;          // center_f
    fm.h[64 + c] = (_Float16)(cv - gv);   // diff_f
  }
  float mx = xyz[(size_t)(b * NN + m) * 3 + 0];
  float my = xyz[(size_t)(b * NN + m) * 3 + 1];
  float mz = xyz[(size_t)(b * NN + m) * 3 + 2];
  float nx = xyz[(size_t)(b * NN + n) * 3 + 0];
  float ny = xyz[(size_t)(b * NN + n) * 3 + 1];
  float nz = xyz[(size_t)(b * NN + n) * 3 + 2];
  float dx = mx - nx, dy = my - ny, dz = mz - nz;
  fm.h[96]  = (_Float16)dx;  fm.h[97]  = (_Float16)dy;  fm.h[98]  = (_Float16)dz;   // xyz_diff
  fm.h[99]  = (_Float16)mx;  fm.h[100] = (_Float16)my;  fm.h[101] = (_Float16)mz;   // grouped_xyz
  fm.h[102] = (_Float16)nx;  fm.h[103] = (_Float16)ny;  fm.h[104] = (_Float16)nz;   // xyz_repeat
  fm.h[105] = (_Float16)sqrtf(dx * dx + dy * dy + dz * dz);                          // xyz_dist
#pragma unroll
  for (int c = CF; c < CFP; ++c) fm.h[c] = (_Float16)0.0f;
  uint4* dst = reinterpret_cast<uint4*>(fmat16 + (size_t)gid * CFP);
#pragma unroll
  for (int i = 0; i < 16; ++i) dst[i] = fm.q[i];
  gxyz[(size_t)gid * 3 + 0] = mx;
  gxyz[(size_t)gid * 3 + 1] = my;
  gxyz[(size_t)gid * 3 + 2] = mz;
}

// ---------------- per-channel gather r = Fmat[..., r_indice] -> r16 [PT*4][128] ----------------
__global__ __launch_bounds__(256) void rgather_kernel(const _Float16* __restrict__ fmat16,
                                                      const int* __restrict__ rind,
                                                      _Float16* __restrict__ r16) {
  int gid = blockIdx.x * 256 + threadIdx.x;            // (b,n,j)
  if (gid >= PT * RR) return;
  int b  = gid >> 15;                                  // / (N*R)
  int nj = gid & (NN * RR - 1);
  int n  = nj >> 2;
  int j  = nj & 3;
  size_t pixBase = (size_t)(b * NN + n) * KK;
  union { _Float16 h[128]; uint4 q[16]; } fm;
#pragma unroll
  for (int c = 0; c < CF; ++c) {
    int k = rind[(((size_t)b * CF + c) * NN + n) * RR + j];
    fm.h[c] = fmat16[(pixBase + k) * CFP + c];
  }
#pragma unroll
  for (int c = CF; c < CFP; ++c) fm.h[c] = (_Float16)0.0f;
  uint4* dst = reinterpret_cast<uint4*>(r16 + (size_t)gid * CFP);
#pragma unroll
  for (int i = 0; i < 16; ++i) dst[i] = fm.q[i];
}

// ---------------- fused g/h/l WMMA GEMM + bias/ReLU + scores + l-max ----------------
__global__ __launch_bounds__(256) void gemm_attn_kernel(const _Float16* __restrict__ fmat16,
                                                        const _Float16* __restrict__ r16,
                                                        const _Float16* __restrict__ w16,
                                                        const float* __restrict__ bg,
                                                        const float* __restrict__ bh,
                                                        const float* __restrict__ bl,
                                                        float* __restrict__ scores,
                                                        unsigned* __restrict__ lprime) {
  const int lane = threadIdx.x & 31;
  const int wave = threadIdx.x >> 5;
  const int hc   = blockIdx.y * 8 + wave;              // h-chunk 0..31 (16 rows each)
  const int l15  = lane & 15;
  const int hi   = lane >> 4;

  // resident A-fragments (weights) for this h-chunk: [mat][kstep]
  const _Float16* wgp = w16 + ((size_t)(0 * HH + hc * 16 + l15)) * CFP;
  const _Float16* whp = w16 + ((size_t)(1 * HH + hc * 16 + l15)) * CFP;
  const _Float16* wlp = w16 + ((size_t)(2 * HH + hc * 16 + l15)) * CFP;
  half16 ag[4], ah[4], al[4];
#pragma unroll
  for (int ks = 0; ks < 4; ++ks) {
    int ho = ks * 32 + hi * 16;
    ag[ks] = load_fragA(wgp + ho);
    ah[ks] = load_fragA(whp + ho);
    al[ks] = load_fragA(wlp + ho);
  }
  float bgv[8], bhv[8], blv[8];
#pragma unroll
  for (int v = 0; v < 8; ++v) {
    int row = hc * 16 + hi * 8 + v;
    bgv[v] = bg[row]; bhv[v] = bh[row]; blv[v] = bl[row];
  }

  for (int pi = 0; pi < PTS_PER_BLOCK; ++pi) {
    const int point = blockIdx.x * PTS_PER_BLOCK + pi;
    const int b = point >> 13;
    const int n = point & (NN - 1);
    const _Float16* brow = fmat16 + ((size_t)point * 16 + l15) * CFP;
    // g's B-matrix: clamp column to 0..3 (duplicate columns 4..15). The duplicated
    // accumulator columns are never read (scores only shuffle lanes r / 16+r, r<4),
    // and this keeps all 32 lanes converged (no exec-mask churn around WMMA).
    const _Float16* grow = r16 + ((size_t)point * 4 + (l15 & 3)) * CFP;
    if (pi + 1 < PTS_PER_BLOCK)
      __builtin_prefetch(fmat16 + ((size_t)(point + 1) * 16 + l15) * CFP, 0, 1);

    float8 ch = {}; float8 cl = {}; float8 cg = {};
#pragma unroll
    for (int ks = 0; ks < 4; ++ks) {
      int ho = ks * 32 + hi * 16;
      half16 bfr = load_fragB(brow + ho);
      half16 gfr = load_fragB(grow + ho);
      ch = __builtin_amdgcn_wmma_f32_16x16x32_f16(false, ah[ks], false, bfr, (short)0, ch, false, false);
      cl = __builtin_amdgcn_wmma_f32_16x16x32_f16(false, al[ks], false, bfr, (short)0, cl, false, false);
      cg = __builtin_amdgcn_wmma_f32_16x16x32_f16(false, ag[ks], false, gfr, (short)0, cg, false, false);
    }

    // bias + ReLU (C layout: vgpr v -> row v (lanes 0-15) / row 8+v (lanes 16-31); lane = col)
    float hr[8], lr[8], gr[8];
#pragma unroll
    for (int v = 0; v < 8; ++v) {
      hr[v] = fmaxf(ch[v] + bhv[v], 0.0f);
      lr[v] = fmaxf(cl[v] + blv[v], 0.0f);
      gr[v] = fmaxf(cg[v] + bgv[v], 0.0f);
    }

    // l' = max over k (cols) -> atomic max (post-ReLU >= 0, uint bit order valid)
#pragma unroll
    for (int v = 0; v < 8; ++v) {
      float mval = lr[v];
      mval = fmaxf(mval, __shfl_xor(mval, 1));
      mval = fmaxf(mval, __shfl_xor(mval, 2));
      mval = fmaxf(mval, __shfl_xor(mval, 4));
      mval = fmaxf(mval, __shfl_xor(mval, 8));
      lr[v] = mval;
    }
    if (l15 == 0) {
#pragma unroll
      for (int v = 0; v < 8; ++v) {
        int row = hc * 16 + hi * 8 + v;
        atomicMax(&lprime[((size_t)b * HH + row) * NN + n], __float_as_uint(lr[v]));
      }
    }

    // scores[r,k] += sum_{rows in this chunk} g[row,r] * h[row,k]
#pragma unroll
    for (int r = 0; r < RR; ++r) {
      int src = hi ? (16 + r) : r;
      float part = 0.0f;
#pragma unroll
      for (int v = 0; v < 8; ++v) part += __shfl(gr[v], src) * hr[v];
      part += __shfl_xor(part, 16);
      if (hi == 0) atomicAdd(&scores[((size_t)point * RR + r) * KK + l15], part);
    }
  }
}

// ---------------- softmax + aff -> coarse_xyz + feature channels 512..514 ----------------
__global__ __launch_bounds__(256) void finalize_kernel(const float* __restrict__ scores,
                                                       const float* __restrict__ gxyz,
                                                       float* __restrict__ out0,
                                                       float* __restrict__ out1) {
  int gid = blockIdx.x * 256 + threadIdx.x;            // (b,n,r)
  if (gid >= PT * RR) return;
  int b  = gid >> 15;
  int nj = gid & (NN * RR - 1);
  int n  = nj >> 2;
  int r  = nj & 3;
  const float* s = scores + (size_t)gid * KK;
  float mx = -3.402823e38f;
#pragma unroll
  for (int k = 0; k < KK; ++k) mx = fmaxf(mx, s[k]);
  float e[KK]; float sum = 0.0f;
#pragma unroll
  for (int k = 0; k < KK; ++k) { e[k] = __expf(s[k] - mx); sum += e[k]; }
  float inv = 1.0f / sum;
  const float* g = gxyz + (size_t)(b * NN + n) * KK * 3;
  float ax = 0.0f, ay = 0.0f, az = 0.0f;
#pragma unroll
  for (int k = 0; k < KK; ++k) {
    float w = e[k] * inv;
    ax += w * g[k * 3 + 0];
    ay += w * g[k * 3 + 1];
    az += w * g[k * 3 + 2];
  }
  size_t co = ((size_t)b * (NN * RR) + n * RR + r) * 3;
  out0[co + 0] = ax; out0[co + 1] = ay; out0[co + 2] = az;
  out1[(((size_t)b * 515 + 512) * NN + n) * RR + r] = ax;
  out1[(((size_t)b * 515 + 513) * NN + n) * RR + r] = ay;
  out1[(((size_t)b * 515 + 514) * NN + n) * RR + r] = az;
}

// ---------------- broadcast l' over R into feature channels 0..511 ----------------
__global__ __launch_bounds__(256) void lbroadcast_kernel(const unsigned* __restrict__ lp,
                                                         float* __restrict__ out1) {
  int gid = blockIdx.x * 256 + threadIdx.x;            // (b,h,n)
  if (gid >= BB * HH * NN) return;
  int b  = gid >> 22;
  int hn = gid & (HH * NN - 1);
  int h  = hn >> 13;
  int n  = hn & (NN - 1);
  float v = __uint_as_float(lp[gid]);
  float4 o = make_float4(v, v, v, v);
  *reinterpret_cast<float4*>(out1 + (((size_t)b * 515 + h) * NN + n) * RR) = o;
}

// ---------------- host launcher ----------------
extern "C" void kernel_launch(void* const* d_in, const int* in_sizes, int n_in,
                              void* d_out, int out_size, void* d_ws, size_t ws_size,
                              hipStream_t stream) {
  const float* xyz  = (const float*)d_in[0];
  const float* f    = (const float*)d_in[1];
  const float* Wg   = (const float*)d_in[2];
  const float* bg   = (const float*)d_in[3];
  const float* Wh   = (const float*)d_in[4];
  const float* bh   = (const float*)d_in[5];
  const float* Wl   = (const float*)d_in[6];
  const float* bl   = (const float*)d_in[7];
  const int*   rind = (const int*)d_in[8];

  char* ws = (char*)d_ws;
  int*        idx    = (int*)(ws + O_IDX);
  float*      gxyz   = (float*)(ws + O_GXYZ);
  _Float16*   fmat16 = (_Float16*)(ws + O_FMAT);
  _Float16*   r16    = (_Float16*)(ws + O_R16);
  _Float16*   w16    = (_Float16*)(ws + O_W16);
  float*      scores = (float*)(ws + O_SCORES);
  unsigned*   lp     = (unsigned*)(ws + O_LP);

  float* out0 = (float*)d_out;                    // coarse_xyz (B, N*R, 3)
  float* out1 = out0 + (size_t)BB * NN * RR * 3;  // output_feature (B, 515, N, R)

  zero_kernel<<<2048, 256, 0, stream>>>((uint4*)(ws + O_SCORES), ZERO_U4);
  wconvert_kernel<<<(3 * 512 * 128) / 256, 256, 0, stream>>>(Wg, Wh, Wl, w16);
  knn_kernel<<<PT / 8, 256, 0, stream>>>(xyz, idx);
  build_kernel<<<PIX / 256, 256, 0, stream>>>(xyz, f, idx, fmat16, gxyz);
  rgather_kernel<<<(PT * RR) / 256, 256, 0, stream>>>(fmat16, rind, r16);
  gemm_attn_kernel<<<dim3(PT / PTS_PER_BLOCK, 4), 256, 0, stream>>>(
      fmat16, r16, w16, bg, bh, bl, scores, lp);
  finalize_kernel<<<(PT * RR) / 256, 256, 0, stream>>>(scores, gxyz, out0, out1);
  lbroadcast_kernel<<<(BB * HH * NN) / 256, 256, 0, stream>>>(lp, out1);
}
